// AqtDotGeneral_8735963480144
// MI455X (gfx1250) — 8-bit matrix operand order repaired and checked by reading, compiled, not run on hardware
//
#include <hip/hip_runtime.h>
#include <stdint.h>

typedef int v8i __attribute__((ext_vector_type(8)));

constexpr int M_DIM = 8192;
constexpr int N_DIM = 4096;
constexpr int K_DIM = 4096;

__global__ __launch_bounds__(256) void quant_lhs_kernel(
    const float* __restrict__ x, int8_t* __restrict__ q, float* __restrict__ s) {
  constexpr int K = K_DIM;
  const int row = blockIdx.x;
  const int t = threadIdx.x;

  const float4* xr = (const float4*)(x + (size_t)row * K);
  float4 v[4];
  float amax = 0.f;
#pragma unroll
  for (int j = 0; j < 4; ++j) {
    v[j] = xr[t * 4 + j];
    amax = fmaxf(amax, fmaxf(fmaxf(fabsf(v[j].x), fabsf(v[j].y)),
                             fmaxf(fabsf(v[j].z), fabsf(v[j].w))));
  }

  __shared__ float red[256];
  red[t] = amax;
  __syncthreads();
#pragma unroll
  for (int off = 128; off > 0; off >>= 1) {
    if (t < off) red[t] = fmaxf(red[t], red[t + off]);
    __syncthreads();
  }
  const float absmax = red[0];
  const float scale = (absmax == 0.f) ? 1.f : absmax * (1.f / 127.5f);
  const float inv = 1.f / scale;
  if (t == 0) s[row] = scale;

  uint32_t w[4];
#pragma unroll
  for (int j = 0; j < 4; ++j) {
    const int q0 = (int)fminf(fmaxf(rintf(v[j].x * inv), -127.f), 127.f);
    const int q1 = (int)fminf(fmaxf(rintf(v[j].y * inv), -127.f), 127.f);
    const int q2 = (int)fminf(fmaxf(rintf(v[j].z * inv), -127.f), 127.f);
    const int q3 = (int)fminf(fmaxf(rintf(v[j].w * inv), -127.f), 127.f);
    w[j] = (uint32_t)(q0 & 255) | ((uint32_t)(q1 & 255) << 8) |
           ((uint32_t)(q2 & 255) << 16) | ((uint32_t)(q3 & 255) << 24);
  }
  int4 st;
  st.x = (int)w[0]; st.y = (int)w[1]; st.z = (int)w[2]; st.w = (int)w[3];
  ((int4*)(q + (size_t)row * K))[t] = st;
}

__global__ __launch_bounds__(256) void quant_rhs_kernel(
    const float* __restrict__ x, int8_t* __restrict__ qT, float* __restrict__ s) {
  constexpr int K = K_DIM, N = N_DIM;
  const int colBase = blockIdx.x * 64;
  const int t = threadIdx.x;
  const int c = t & 63;
  const int kq = t >> 6;

  float amax = 0.f;
  for (int k = kq; k < K; k += 4)
    amax = fmaxf(amax, fabsf(x[(size_t)k * N + colBase + c]));

  __shared__ float part[4][64];
  __shared__ float sinv[64];
  part[kq][c] = amax;
  __syncthreads();
  if (t < 64) {
    const float am = fmaxf(fmaxf(part[0][t], part[1][t]),
                           fmaxf(part[2][t], part[3][t]));
    const float scale = (am == 0.f) ? 1.f : am * (1.f / 127.5f);
    s[colBase + t] = scale;
    sinv[t] = 1.f / scale;
  }
  __syncthreads();

  __shared__ char tile[64 * 80];
  const int c2 = t >> 2;
  const int seg = t & 3;
  for (int kt = 0; kt < K; kt += 64) {
#pragma unroll 4
    for (int i = 0; i < 16; ++i) {
      const int idx = t + 256 * i;
      const int r = idx >> 6;
      const int cc = idx & 63;
      const float xv = x[(size_t)(kt + r) * N + colBase + cc];
      const int qv = (int)fminf(fmaxf(rintf(xv * sinv[cc]), -127.f), 127.f);
      tile[cc * 80 + r] = (char)qv;
    }
    __syncthreads();
    const int4 val = *(const int4*)&tile[c2 * 80 + seg * 16];
    *(int4*)(qT + (size_t)(colBase + c2) * K + kt + seg * 16) = val;
    __syncthreads();
  }
}

__global__ __launch_bounds__(256) void gemm_q8_kernel(
    const int8_t* __restrict__ qA,
    const int8_t* __restrict__ qB,
    const float* __restrict__ sA,
    const float* __restrict__ sB,
    float* __restrict__ out) {
  constexpr int K = K_DIM, N = N_DIM;
  constexpr int BM = 128, BN = 128, BK = 64;
  constexpr int LDA = 80, LDB = 80;

  __shared__ char As[BM * LDA];
  __shared__ char Bs[BN * LDB];

  const int t = threadIdx.x;
  const int lane = t & 31;
  const int wave = t >> 5;
  const int waveM = wave & 3;
  const int waveN = wave >> 2;
  const int l16 = lane & 15;
  const int half = lane >> 4;

  const int bM = blockIdx.y * BM;
  const int bN = blockIdx.x * BN;

  const int sRow = t >> 1;
  const int sSeg = (t & 1) * 32;
  const int8_t* gA = qA + (size_t)(bM + sRow) * K + sSeg;
  const int8_t* gB = qB + (size_t)(bN + sRow) * K + sSeg;
  char* lA = &As[sRow * LDA + sSeg];
  char* lB = &Bs[sRow * LDB + sSeg];

  v8i acc[2][4] = {};

  for (int kt = 0; kt < K; kt += BK) {
    const int4 a0 = *(const int4*)(gA + kt);
    const int4 a1 = *(const int4*)(gA + kt + 16);
    const int4 b0 = *(const int4*)(gB + kt);
    const int4 b1 = *(const int4*)(gB + kt + 16);
    __syncthreads();
    *(int4*)lA = a0;
    *(int4*)(lA + 16) = a1;
    *(int4*)lB = b0;
    *(int4*)(lB + 16) = b1;
    __syncthreads();
    if (kt + BK < K) {
      __builtin_prefetch(gA + kt + BK, 0, 0);
      __builtin_prefetch(gB + kt + BK, 0, 0);
    }

    v8i af[2];
#pragma unroll
    for (int mi = 0; mi < 2; ++mi) {
      const char* ap = &As[(waveM * 32 + mi * 16 + l16) * LDA + half * 16];
      const int2 c0 = *(const int2*)(ap);
      const int2 c1 = *(const int2*)(ap + 8);
      const int2 c2 = *(const int2*)(ap + 32);
      const int2 c3 = *(const int2*)(ap + 40);
      v8i av;
      av[0] = c0.x; av[1] = c0.y; av[2] = c1.x; av[3] = c1.y;
      av[4] = c2.x; av[5] = c2.y; av[6] = c3.x; av[7] = c3.y;
      af[mi] = av;
    }
    v8i bf[4];
#pragma unroll
    for (int ni = 0; ni < 4; ++ni) {
      const char* bp = &Bs[(waveN * 64 + ni * 16 + l16) * LDB + half * 16];
      const int4 b0v = *(const int4*)(bp);
      const int4 b1v = *(const int4*)(bp + 32);
      v8i bv;
      bv[0] = b0v.x; bv[1] = b0v.y; bv[2] = b0v.z; bv[3] = b0v.w;
      bv[4] = b1v.x; bv[5] = b1v.y; bv[6] = b1v.z; bv[7] = b1v.w;
      bf[ni] = bv;
    }

#pragma unroll
    for (int mi = 0; mi < 2; ++mi)
#pragma unroll
      for (int ni = 0; ni < 4; ++ni)
        acc[mi][ni] = __builtin_amdgcn_wmma_i32_16x16x64_iu8(
             true, af[mi],  true, bf[ni], acc[mi][ni],
             false,  false);
  }

#pragma unroll
  for (int mi = 0; mi < 2; ++mi) {
    const int mBase = bM + waveM * 32 + mi * 16 + half * 8;
    float sl[8];
#pragma unroll
    for (int r = 0; r < 8; ++r) sl[r] = sA[mBase + r];
#pragma unroll
    for (int ni = 0; ni < 4; ++ni) {
      const int gn = bN + waveN * 64 + ni * 16 + l16;
      const float sr = sB[gn];
#pragma unroll
      for (int r = 0; r < 8; ++r)
        out[(size_t)(mBase + r) * N + gn] = (float)acc[mi][ni][r] * sl[r] * sr;
    }
  }
}

extern "C" void kernel_launch(void* const* d_in, const int* in_sizes, int n_in,
                              void* d_out, int out_size, void* d_ws, size_t ws_size,
                              hipStream_t stream) {
  const float* lhs = (const float*)d_in[0];
  const float* rhs = (const float*)d_in[1];
  float* out = (float*)d_out;

  int8_t* q_lhs = (int8_t*)d_ws;
  int8_t* q_rhsT = q_lhs + (size_t)M_DIM * K_DIM;
  float* s_lhs = (float*)(q_rhsT + (size_t)N_DIM * K_DIM);
  float* s_rhs = s_lhs + M_DIM;

  quant_lhs_kernel<<<M_DIM, 256, 0, stream>>>(lhs, q_lhs, s_lhs);
  quant_rhs_kernel<<<N_DIM / 64, 256, 0, stream>>>(rhs, q_rhsT, s_rhs);
  gemm_q8_kernel<<<dim3(N_DIM / 128, M_DIM / 128), 256, 0, stream>>>(
      q_lhs, q_rhsT, s_lhs, s_rhs, out);
}
